// Two_Hop_GAT_37606733643856
// MI455X (gfx1250) — matrix operand order repaired and checked by reading, not compiled, not run on hardware
//
#include <hip/hip_runtime.h>
#include <hip/hip_bf16.h>

// ---------------------------------------------------------------------------
// Two-hop GAT for MI455X (gfx1250): bf16 WMMA GEMMs (pipelined) +
// CSR-gather edge softmax/aggregation (no float atomics)
// ---------------------------------------------------------------------------

#define NEG_SLOPE 0.2f
#define EPS_F 1e-16f

typedef __bf16 bf16_t;
typedef bf16_t bf16x16 __attribute__((ext_vector_type(16)));
typedef float f32x8 __attribute__((ext_vector_type(8)));

union Frag32B { bf16x16 v; uint4 q[2]; };

__device__ __forceinline__ unsigned short f32_to_bf16_rn(float f) {
  unsigned int u = __float_as_uint(f);
  if ((u & 0x7fffffffu) > 0x7f800000u) return (unsigned short)((u >> 16) | 0x0040u);
  unsigned int lsb = (u >> 16) & 1u;
  u += 0x7fffu + lsb;
  return (unsigned short)(u >> 16);
}

// ----------------------------- elementwise ---------------------------------

__global__ void k_f32_to_bf16(const float* __restrict__ in,
                              unsigned short* __restrict__ out, long n) {
  long i = (long)blockIdx.x * blockDim.x + threadIdx.x;
  if (i < n) out[i] = f32_to_bf16_rn(in[i]);
}

// W [K, Nc] f32 row-major  ->  Wt [Nc, K] bf16 (i.e. B^T row-major)
__global__ void k_w_to_bf16_t(const float* __restrict__ W,
                              unsigned short* __restrict__ Wt, int K, int Nc) {
  int t = blockIdx.x * blockDim.x + threadIdx.x;
  if (t < K * Nc) {
    int k = t / Nc, n = t % Nc;
    Wt[(long)n * K + k] = f32_to_bf16_rn(W[t]);
  }
}

__global__ void k_fill_i32(int* __restrict__ p, int v, long n) {
  long i = (long)blockIdx.x * blockDim.x + threadIdx.x;
  if (i < n) p[i] = v;
}

__global__ void k_copy_i32(const int* __restrict__ a, int* __restrict__ b, long n) {
  long i = (long)blockIdx.x * blockDim.x + threadIdx.x;
  if (i < n) b[i] = a[i];
}

// acc[i, c] = 2 * bias[c]   (two gat_conv calls each add bias once)
__global__ void k_init_bias2(float* __restrict__ out, const float* __restrict__ b,
                             long n, int C) {
  long i = (long)blockIdx.x * blockDim.x + threadIdx.x;
  if (i < n) out[i] = 2.0f * b[i % C];
}

// ------------------------ small dots (alpha path) ---------------------------

// out[k] = sum_c W[k, c] * a[c]     (W: [K, C] f32)  -- one wave per row
__global__ void k_w_dot_a(const float* __restrict__ W, const float* __restrict__ a,
                          float* __restrict__ out, int K, int C) {
  int row  = blockIdx.x * 8 + (threadIdx.x >> 5);
  int lane = threadIdx.x & 31;
  if (row >= K) return;
  const float* wr = W + (long)row * C;
  float v = 0.f;
  for (int c = lane; c < C; c += 32) v += wr[c] * a[c];
  #pragma unroll
  for (int off = 16; off > 0; off >>= 1) v += __shfl_down(v, off, 32);
  if (lane == 0) out[row] = v;
}

// alpha[i] = sum_c X[i, c] * w[c]   (X: [N, Din] f32) -- one wave per node
__global__ void k_node_dot(const float* __restrict__ X, const float* __restrict__ w,
                           float* __restrict__ out, int Nn, int Din) {
  int node = blockIdx.x * 8 + (threadIdx.x >> 5);
  int lane = threadIdx.x & 31;
  if (node >= Nn) return;
  const float* xr = X + (long)node * Din;
  float v = 0.f;
  for (int c = lane; c < Din; c += 32) v += xr[c] * w[c];
  #pragma unroll
  for (int off = 16; off > 0; off >>= 1) v += __shfl_down(v, off, 32);
  if (lane == 0) out[node] = v;
}

// ------------------------------ CSR build -----------------------------------

__global__ void k_count(const int* __restrict__ dst, int* __restrict__ deg, int En) {
  int i = blockIdx.x * blockDim.x + threadIdx.x;
  if (i < En) atomicAdd(&deg[dst[i]], 1);
}

// single-block exclusive scan: ofs[0..Nn] from deg[0..Nn)
__global__ void __launch_bounds__(1024)
k_scan_offsets(const int* __restrict__ deg, int* __restrict__ ofs, int Nn) {
  __shared__ int part[1024];
  int t = threadIdx.x;
  int chunk = (Nn + 1023) >> 10;
  int beg = t * chunk;
  int end = beg + chunk; if (end > Nn) end = Nn;
  int s = 0;
  for (int i = beg; i < end; ++i) s += deg[i];
  part[t] = s;
  __syncthreads();
  if (t == 0) {
    int run = 0;
    for (int i = 0; i < 1024; ++i) { int v = part[i]; part[i] = run; run += v; }
    ofs[Nn] = run;   // total
  }
  __syncthreads();
  int run = part[t];
  for (int i = beg; i < end; ++i) { ofs[i] = run; run += deg[i]; }
}

__global__ void k_place(const int* __restrict__ dst, int* __restrict__ cursor,
                        int* __restrict__ eid, int En) {
  int i = blockIdx.x * blockDim.x + threadIdx.x;
  if (i < En) {
    int p = atomicAdd(&cursor[dst[i]], 1);
    eid[p] = i;
  }
}

// ------------------------- segmented softmax (gather) -----------------------
// one wave per destination node; alpha stored per CSR position
__global__ void k_node_softmax(const int* __restrict__ eid, const int* __restrict__ ofs,
                               const int* __restrict__ src,
                               const float* __restrict__ aS, const float* __restrict__ aD,
                               float* __restrict__ alpha, int Nn) {
  int d    = blockIdx.x * 8 + (threadIdx.x >> 5);
  int lane = threadIdx.x & 31;
  if (d >= Nn) return;
  int b = ofs[d], e = ofs[d + 1];
  if (b == e) return;
  float ad = aD[d];

  float mx = -3.0e38f;
  for (int j = b + lane; j < e; j += 32) {
    float l = aS[src[eid[j]]] + ad;
    l = (l > 0.f) ? l : NEG_SLOPE * l;     // leaky_relu
    alpha[j] = l;
    mx = fmaxf(mx, l);
  }
  #pragma unroll
  for (int off = 16; off > 0; off >>= 1) mx = fmaxf(mx, __shfl_xor(mx, off, 32));

  float s = 0.f;
  for (int j = b + lane; j < e; j += 32) {
    float ex = expf(alpha[j] - mx);
    alpha[j] = ex;
    s += ex;
  }
  #pragma unroll
  for (int off = 16; off > 0; off >>= 1) s += __shfl_xor(s, off, 32);

  float inv = 1.f / (s + EPS_F);
  for (int j = b + lane; j < e; j += 32) alpha[j] *= inv;
}

// ---------------------- per-destination gather (no atomics) -----------------
// one block per destination node: acc[d,:] += sum_j alpha[j] * h_src[src_j,:]
__global__ void __launch_bounds__(256)
k_node_gather(const int* __restrict__ eid, const int* __restrict__ ofs,
              const int* __restrict__ src, const float* __restrict__ alpha,
              const float* __restrict__ hs, float* __restrict__ acc, int C) {
  int d = blockIdx.x;
  int b = ofs[d], e = ofs[d + 1];
  if (b == e) return;
  __shared__ int   ssrc[64];
  __shared__ float sal[64];
  for (int base = b; base < e; base += 64) {
    int cnt = e - base; if (cnt > 64) cnt = 64;
    if ((int)threadIdx.x < cnt) {
      ssrc[threadIdx.x] = src[eid[base + threadIdx.x]];
      sal[threadIdx.x]  = alpha[base + threadIdx.x];
    }
    __syncthreads();
    for (int c = threadIdx.x; c < C; c += blockDim.x) {
      float v = 0.f;
      for (int j = 0; j < cnt; ++j) v += sal[j] * hs[(long)ssrc[j] * C + c];
      acc[(long)d * C + c] += v;
    }
    __syncthreads();
  }
}

// ------------------------------ WMMA GEMM -----------------------------------
// out[M, Nc] (f32) = A[M, K] (bf16, row-major) x Bt[Nc, K]^T (bf16) [+bias][relu]
// wave tile 16x64, 8 waves/block = 128x64; software-pipelined fragment loads.
__global__ void __launch_bounds__(256)
k_gemm_bf16(const unsigned short* __restrict__ A, const unsigned short* __restrict__ Bt,
            float* __restrict__ out, const float* __restrict__ bias,
            int relu, int M, int K, int Nc) {
  const int lane = threadIdx.x & 31;
  const int wave = threadIdx.x >> 5;
  const int m0   = blockIdx.y * 128 + wave * 16;
  const int n0   = blockIdx.x * 64;
  const int half = lane >> 4;   // lane group (0: lanes 0-15, 1: lanes 16-31)
  const int l16  = lane & 15;

  // A-fragment lane layout (ISA 7.12.2, 16-bit A 16x32)
  int arow = m0 + l16; if (arow >= M) arow = M - 1;   // branchless clamp (EXEC all-1s)
  const unsigned short* aptr = A + (size_t)arow * K + half * 8;
  const unsigned short* bp[4];
  #pragma unroll
  for (int j = 0; j < 4; ++j)
    bp[j] = Bt + (size_t)(n0 + j * 16 + l16) * K + half * 8;

  f32x8 acc[4] = {{0.f,0.f,0.f,0.f,0.f,0.f,0.f,0.f},
                  {0.f,0.f,0.f,0.f,0.f,0.f,0.f,0.f},
                  {0.f,0.f,0.f,0.f,0.f,0.f,0.f,0.f},
                  {0.f,0.f,0.f,0.f,0.f,0.f,0.f,0.f}};

  // prologue: fragments for kk = 0
  Frag32B fa, fb[4];
  fa.q[0] = *(const uint4*)(aptr);
  fa.q[1] = *(const uint4*)(aptr + 16);
  #pragma unroll
  for (int j = 0; j < 4; ++j) {
    fb[j].q[0] = *(const uint4*)(bp[j]);
    fb[j].q[1] = *(const uint4*)(bp[j] + 16);
  }

  for (int kk = 32; kk < K; kk += 32) {
    // issue next iteration's loads before consuming current fragments
    Frag32B na, nb[4];
    na.q[0] = *(const uint4*)(aptr + kk);
    na.q[1] = *(const uint4*)(aptr + kk + 16);
    #pragma unroll
    for (int j = 0; j < 4; ++j) {
      nb[j].q[0] = *(const uint4*)(bp[j] + kk);
      nb[j].q[1] = *(const uint4*)(bp[j] + kk + 16);
    }
    #pragma unroll
    for (int j = 0; j < 4; ++j)
      acc[j] = __builtin_amdgcn_wmma_f32_16x16x32_bf16(
          false, fa.v, false, fb[j].v, (short)0, acc[j], false, false);
    fa = na;
    #pragma unroll
    for (int j = 0; j < 4; ++j) fb[j] = nb[j];
  }
  #pragma unroll
  for (int j = 0; j < 4; ++j)
    acc[j] = __builtin_amdgcn_wmma_f32_16x16x32_bf16(
        false, fa.v, false, fb[j].v, (short)0, acc[j], false, false);

  // C/D layout: VGPR r -> row m0 + half*8 + r, col = subtile + l16
  #pragma unroll
  for (int j = 0; j < 4; ++j) {
    int col = n0 + j * 16 + l16;
    float bv = bias ? bias[col] : 0.f;
    #pragma unroll
    for (int r = 0; r < 8; ++r) {
      int row = m0 + half * 8 + r;
      if (row < M) {
        float v = acc[j][r] + bv;
        if (relu) v = fmaxf(v, 0.f);
        out[(size_t)row * Nc + col] = v;
      }
    }
  }
}

// ------------------------------- host side ----------------------------------

static inline long cdivl(long a, long b) { return (a + b - 1) / b; }

extern "C" void kernel_launch(void* const* d_in, const int* in_sizes, int n_in,
                              void* d_out, int out_size, void* d_ws, size_t ws_size,
                              hipStream_t stream) {
  const int D = 512, H = 512, O = 256;
  const int N = in_sizes[0] / D;
  const int E = in_sizes[1] / 2;

  const float* x    = (const float*)d_in[0];
  const int*   ei1  = (const int*)d_in[1];   // [0..E): src, [E..2E): dst
  const int*   ei2  = (const int*)d_in[2];
  const float* W1s  = (const float*)d_in[3];
  const float* W1d  = (const float*)d_in[4];
  const float* a1s  = (const float*)d_in[5];
  const float* a1d  = (const float*)d_in[6];
  const float* b1   = (const float*)d_in[7];
  const float* Wl1  = (const float*)d_in[8];
  const float* bl1  = (const float*)d_in[9];
  const float* W2s  = (const float*)d_in[10];
  const float* W2d  = (const float*)d_in[11];
  const float* a2s  = (const float*)d_in[12];
  const float* a2d  = (const float*)d_in[13];
  const float* b2   = (const float*)d_in[14];
  const float* Wl2  = (const float*)d_in[15];
  const float* bl2  = (const float*)d_in[16];
  float* out = (float*)d_out;

  // ---- workspace carve ----
  char* ws = (char*)d_ws; size_t off = 0;
  auto alloc = [&](size_t bytes) -> void* {
    off = (off + 255) & ~(size_t)255;
    void* p = ws + off; off += bytes; return p;
  };
  unsigned short* xbf  = (unsigned short*)alloc((size_t)N * D * 2); // bf16 layer input
  unsigned short* habf = (unsigned short*)alloc((size_t)N * H * 2); // bf16 of conv accum
  float* bufA = (float*)alloc((size_t)N * H * 4);  // h_src / lin / h_src2 (aliased)
  float* bufB = (float*)alloc((size_t)N * H * 4);  // conv accumulator
  unsigned short* Wt1s = (unsigned short*)alloc((size_t)H * D * 2);
  unsigned short* Wtl1 = (unsigned short*)alloc((size_t)H * H * 2);
  unsigned short* Wt2s = (unsigned short*)alloc((size_t)O * H * 2);
  unsigned short* Wtl2 = (unsigned short*)alloc((size_t)O * O * 2);
  float* wa1s = (float*)alloc((size_t)D * 4);
  float* wa1d = (float*)alloc((size_t)D * 4);
  float* wa2s = (float*)alloc((size_t)H * 4);
  float* wa2d = (float*)alloc((size_t)H * 4);
  float* alS  = (float*)alloc((size_t)N * 4);
  float* alD  = (float*)alloc((size_t)N * 4);
  int*   degB = (int*)alloc((size_t)N * 4);        // CSR degree
  int*   ofsB = (int*)alloc((size_t)(N + 1) * 4);  // CSR row offsets
  int*   curB = (int*)alloc((size_t)N * 4);        // placement cursor
  int*   eidB = (int*)alloc((size_t)E * 4);        // CSR edge ids
  float* alB  = (float*)alloc((size_t)E * 4);      // per-position alpha
  (void)ws_size; (void)n_in; (void)out_size;

  const int T = 256;
  auto gemm = [&](const unsigned short* A, const unsigned short* Bt, float* o,
                  const float* bias, int relu, int M, int K, int Nc) {
    dim3 g(Nc / 64, (unsigned)cdivl(M, 128));
    k_gemm_bf16<<<g, T, 0, stream>>>(A, Bt, o, bias, relu, M, K, Nc);
  };
  auto run_edges = [&](const int* ei, const float* aS, const float* aD,
                       const float* hs, float* acc, int C) {
    const int* src = ei; const int* dst = ei + E;
    k_fill_i32<<<cdivl(N, T), T, 0, stream>>>(degB, 0, N);
    k_count<<<cdivl(E, T), T, 0, stream>>>(dst, degB, E);
    k_scan_offsets<<<1, 1024, 0, stream>>>(degB, ofsB, N);
    k_copy_i32<<<cdivl(N, T), T, 0, stream>>>(ofsB, curB, N);
    k_place<<<cdivl(E, T), T, 0, stream>>>(dst, curB, eidB, E);
    k_node_softmax<<<cdivl(N, 8), T, 0, stream>>>(eidB, ofsB, src, aS, aD, alB, N);
    k_node_gather<<<N, T, 0, stream>>>(eidB, ofsB, src, alB, hs, acc, C);
  };

  // ============================= Layer 1 =============================
  k_f32_to_bf16<<<cdivl((long)N * D, T), T, 0, stream>>>(x, xbf, (long)N * D);
  k_w_to_bf16_t<<<cdivl((long)D * H, T), T, 0, stream>>>(W1s, Wt1s, D, H);
  k_w_dot_a<<<cdivl(D, 8), T, 0, stream>>>(W1s, a1s, wa1s, D, H);
  k_w_dot_a<<<cdivl(D, 8), T, 0, stream>>>(W1d, a1d, wa1d, D, H);

  gemm(xbf, Wt1s, bufA, nullptr, 0, N, D, H);                      // h_src1 = X @ W1_src
  k_node_dot<<<cdivl(N, 8), T, 0, stream>>>(x, wa1s, alS, N, D);   // alpha_src = X @ (W a)
  k_node_dot<<<cdivl(N, 8), T, 0, stream>>>(x, wa1d, alD, N, D);   // alpha_dst

  k_init_bias2<<<cdivl((long)N * H, T), T, 0, stream>>>(bufB, b1, (long)N * H, H);
  run_edges(ei1, alS, alD, bufA, bufB, H);                         // hop 1
  run_edges(ei2, alS, alD, bufA, bufB, H);                         // hop 2

  // lin1 + relu  (writes bufA; h_src1 is dead now)
  k_f32_to_bf16<<<cdivl((long)N * H, T), T, 0, stream>>>(bufB, habf, (long)N * H);
  k_w_to_bf16_t<<<cdivl((long)H * H, T), T, 0, stream>>>(Wl1, Wtl1, H, H);
  gemm(habf, Wtl1, bufA, bl1, 1, N, H, H);                         // h = relu(acc @ Wl1 + bl1)

  // ============================= Layer 2 =============================
  k_f32_to_bf16<<<cdivl((long)N * H, T), T, 0, stream>>>(bufA, xbf, (long)N * H);
  k_w_to_bf16_t<<<cdivl((long)H * O, T), T, 0, stream>>>(W2s, Wt2s, H, O);
  k_w_dot_a<<<cdivl(H, 8), T, 0, stream>>>(W2s, a2s, wa2s, H, O);
  k_w_dot_a<<<cdivl(H, 8), T, 0, stream>>>(W2d, a2d, wa2d, H, O);

  // alphas must read f32 h (bufA) BEFORE it is overwritten by h_src2
  k_node_dot<<<cdivl(N, 8), T, 0, stream>>>(bufA, wa2s, alS, N, H);
  k_node_dot<<<cdivl(N, 8), T, 0, stream>>>(bufA, wa2d, alD, N, H);
  gemm(xbf, Wt2s, bufA, nullptr, 0, N, H, O);                      // h_src2 (overwrites lin1)

  k_init_bias2<<<cdivl((long)N * O, T), T, 0, stream>>>(bufB, b2, (long)N * O, O);
  run_edges(ei1, alS, alD, bufA, bufB, O);
  run_edges(ei2, alS, alD, bufA, bufB, O);

  // lin2 -> d_out
  k_f32_to_bf16<<<cdivl((long)N * O, T), T, 0, stream>>>(bufB, habf, (long)N * O);
  k_w_to_bf16_t<<<cdivl((long)O * O, T), T, 0, stream>>>(Wl2, Wtl2, O, O);
  gemm(habf, Wtl2, out, bl2, 0, N, O, O);
}
